// GNN_LSTM_65163243815593
// MI455X (gfx1250) — matrix operand order repaired and checked by reading, not compiled, not run on hardware
//
#include <hip/hip_runtime.h>
#include <math.h>

// Problem constants (match the reference).
#define B_    2
#define S_    32
#define N_    1024
#define E_    16384
#define D_    64
#define G_    (B_ * S_)        // 64 graphs
#define NTOT_ (G_ * N_)        // 65536 nodes

typedef __attribute__((ext_vector_type(16))) _Float16 v16h;
typedef __attribute__((ext_vector_type(8)))  float    v8f;

__device__ __forceinline__ float sigmoid_(float x) { return 1.0f / (1.0f + expf(-x)); }

// ---------------------------------------------------------------------------
// 1) Degree / normalization. Edge pattern identical across the 64 graphs, so
//    compute once for N=1024 nodes. deg[n] = (#edges with col==n) + 1 (self-loop).
// ---------------------------------------------------------------------------
__global__ void __launch_bounds__(1024) k_degree(const long long* __restrict__ ei,
                                                 float* __restrict__ dis) {
    __shared__ int cnt[N_];
    int tid = threadIdx.x;
    cnt[tid] = 1;                       // self loop
    __syncthreads();
    for (int e = tid; e < E_; e += 1024) {
        int c = (int)ei[E_ + e];        // edge_index[1][e]
        atomicAdd(&cnt[c], 1);
    }
    __syncthreads();
    dis[tid] = rsqrtf((float)cnt[tid]);
}

// ---------------------------------------------------------------------------
// 2) Layer-1 transform: u = (x @ W1) * dis[node].  K=3 -> plain VALU.
// ---------------------------------------------------------------------------
__global__ void k_xform1(const float* __restrict__ x, const float* __restrict__ W1,
                         const float* __restrict__ dis, float* __restrict__ u) {
    int idx  = blockIdx.x * blockDim.x + threadIdx.x;   // NTOT_*64 threads
    int node = idx >> 6;
    int f    = idx & 63;
    const float* xr = x + node * 3;
    float s = xr[0] * W1[f] + xr[1] * W1[64 + f] + xr[2] * W1[128 + f];
    u[idx] = s * dis[node & (N_ - 1)];
}

// ---------------------------------------------------------------------------
// 3) acc init = u (this IS the self-loop term, since norm_ii = dis_i^2).
// ---------------------------------------------------------------------------
__global__ void k_copy4(const float4* __restrict__ src, float4* __restrict__ dst) {
    int i = blockIdx.x * blockDim.x + threadIdx.x;
    dst[i] = src[i];
}

// ---------------------------------------------------------------------------
// 4) Scatter: acc[col] += u[row] for every (graph, edge), 64 features/edge.
//    256 threads = 4 edge-instances x 64 features (coalesced atomics).
// ---------------------------------------------------------------------------
__global__ void __launch_bounds__(256) k_scatter(const long long* __restrict__ ei,
                                                 const float* __restrict__ u,
                                                 float* __restrict__ acc) {
    int t    = blockIdx.x * 256 + threadIdx.x;
    int inst = t >> 6;                 // edge instance in [0, G_*E_)
    int f    = t & 63;
    int g    = inst >> 14;             // / E_  (E_ = 2^14)
    int e    = inst & (E_ - 1);
    if (f == 0) __builtin_prefetch(&ei[e + 256], 0, 0);   // global_prefetch_b8
    int r    = (int)ei[e];
    int c    = (int)ei[E_ + e];
    int base = g * N_;
    atomicAdd(&acc[(base + c) * 64 + f], u[(base + r) * 64 + f]);
}

// ---------------------------------------------------------------------------
// 5) Finish a GCN layer: h = relu(dis[node] * acc + b[f]).
// ---------------------------------------------------------------------------
__global__ void k_finish(const float* __restrict__ acc, const float* __restrict__ dis,
                         const float* __restrict__ b, float* __restrict__ h) {
    int idx  = blockIdx.x * blockDim.x + threadIdx.x;
    int node = idx >> 6;
    int f    = idx & 63;
    float v  = dis[node & (N_ - 1)] * acc[idx] + b[f];
    h[idx]   = v > 0.0f ? v : 0.0f;
}

// ---------------------------------------------------------------------------
// 6) Layer-2 transform with WMMA: u2 = (h1 @ W2) * dis.
//    One wave computes a 16-node x 64-col tile; K=64 via two
//    v_wmma_f32_16x16x32_f16. 4 waves/block -> 64 nodes per block.
//    Lane layouts per CDNA5 ISA 7.12.2:
//      A (16x32 f16): lane<16 -> M=lane,   elems 0..7 = K 0..7,  8..15 = K 16..23
//                     lane>=16 -> M=lane-16, elems 0..7 = K 8..15, 8..15 = K 24..31
//      B (32x16 f16): N = lane&15, elem i = same K as A's elem i
//      C/D (f32):     row = r + (lane<16 ? 0 : 8), col = lane&15
// ---------------------------------------------------------------------------
__global__ void __launch_bounds__(128) k_xform2_wmma(const float* __restrict__ h,
                                                     const float* __restrict__ W2,
                                                     const float* __restrict__ dis,
                                                     float* __restrict__ u2) {
    int lane     = threadIdx.x & 31;
    int wave     = threadIdx.x >> 5;
    int nodeBase = blockIdx.x * 64 + wave * 16;

    int m     = lane & 15;
    int koffA = (lane < 16) ? 0 : 8;
    const float* arow = h + (nodeBase + m) * 64;

    v16h a0, a1;
#pragma unroll
    for (int i = 0; i < 8; ++i) {
        a0[i]     = (_Float16)arow[koffA + i];
        a0[i + 8] = (_Float16)arow[koffA + 16 + i];
        a1[i]     = (_Float16)arow[32 + koffA + i];
        a1[i + 8] = (_Float16)arow[32 + koffA + 16 + i];
    }

    int n     = lane & 15;
    int koffB = (lane < 16) ? 0 : 8;
    int rbase = (lane < 16) ? 0 : 8;

#pragma unroll
    for (int nt = 0; nt < 4; ++nt) {
        int ncol = nt * 16 + n;
        v16h b0, b1;
#pragma unroll
        for (int i = 0; i < 16; ++i) {
            b0[i] = (_Float16)W2[(koffB + i + (i >= 8 ? 8 : 0)) * 64 + ncol];
            b1[i] = (_Float16)W2[(32 + koffB + i + (i >= 8 ? 8 : 0)) * 64 + ncol];
        }
        v8f c = {};
        c = __builtin_amdgcn_wmma_f32_16x16x32_f16(false, a0, false, b0,
                                                   (short)0, c, false, false);
        c = __builtin_amdgcn_wmma_f32_16x16x32_f16(false, a1, false, b1,
                                                   (short)0, c, false, false);
#pragma unroll
        for (int r = 0; r < 8; ++r) {
            int node = nodeBase + rbase + r;
            u2[node * 64 + ncol] = c[r] * dis[node & (N_ - 1)];
        }
    }
}

// ---------------------------------------------------------------------------
// 7) Mean over the 1024 nodes of each graph -> seq (B*S, 64).
// ---------------------------------------------------------------------------
__global__ void __launch_bounds__(64) k_mean(const float* __restrict__ h2,
                                             float* __restrict__ seq) {
    int g = blockIdx.x;      // 0..63
    int d = threadIdx.x;     // 0..63
    const float* p = h2 + g * N_ * 64 + d;
    float s = 0.0f;
    for (int nidx = 0; nidx < N_; ++nidx) s += p[nidx * 64];
    seq[g * 64 + d] = s * (1.0f / (float)N_);
}

// ---------------------------------------------------------------------------
// 8) Both LSTM layers + head, single workgroup (256 threads, 8 waves).
//    xg (input gates) precomputed per layer into global ws; h/c/hs live in LDS.
//    Gate order i,f,g,o per jnp.split.
// ---------------------------------------------------------------------------
__global__ void __launch_bounds__(256) k_lstm(
    const float* __restrict__ seq,
    const float* __restrict__ wih0, const float* __restrict__ whh0,
    const float* __restrict__ bih0, const float* __restrict__ bhh0,
    const float* __restrict__ wih1, const float* __restrict__ whh1,
    const float* __restrict__ bih1, const float* __restrict__ bhh1,
    const float* __restrict__ Wh,   const float* __restrict__ bh,
    float* __restrict__ xg, float* __restrict__ out) {

    __shared__ float sh_h[B_][D_];
    __shared__ float sh_c[B_][D_];
    __shared__ float sh_g[B_][4 * D_];
    __shared__ float sh_hs[B_ * S_ * D_];   // layer-0 outputs (16 KB)

    int tid = threadIdx.x;

    for (int layer = 0; layer < 2; ++layer) {
        const float* wih = layer ? wih1 : wih0;
        const float* whh = layer ? whh1 : whh0;
        const float* bi  = layer ? bih1 : bih0;
        const float* bhv = layer ? bhh1 : bhh0;
        const float* inp = layer ? (const float*)sh_hs : seq;

        // xg[b,s,g] = inp[b,s,:] . wih[g,:] + b_ih[g] + b_hh[g]
        for (int idx = tid; idx < B_ * S_ * 4 * D_; idx += 256) {
            int gate = idx & 255;
            int bs   = idx >> 8;
            const float* xp = inp + bs * D_;
            const float* wp = wih + gate * D_;
            float s = bi[gate] + bhv[gate];
            for (int j = 0; j < D_; ++j) s += xp[j] * wp[j];
            xg[idx] = s;
        }
        if (tid < B_ * D_) {
            sh_h[tid >> 6][tid & 63] = 0.0f;
            sh_c[tid >> 6][tid & 63] = 0.0f;
        }
        __syncthreads();

        for (int t = 0; t < S_; ++t) {
            int gate = tid;  // 0..255
            float g0 = xg[(0 * S_ + t) * 256 + gate];
            float g1 = xg[(1 * S_ + t) * 256 + gate];
            const float* wp = whh + gate * D_;
            for (int j = 0; j < D_; ++j) {
                float w = wp[j];
                g0 += sh_h[0][j] * w;
                g1 += sh_h[1][j] * w;
            }
            __syncthreads();           // protect sh_h reads above from writes below
            sh_g[0][gate] = g0;
            sh_g[1][gate] = g1;
            __syncthreads();
            if (tid < B_ * D_) {
                int b = tid >> 6, j = tid & 63;
                float iv = sigmoid_(sh_g[b][j]);
                float fv = sigmoid_(sh_g[b][64 + j]);
                float gv = tanhf(sh_g[b][128 + j]);
                float ov = sigmoid_(sh_g[b][192 + j]);
                float cc = fv * sh_c[b][j] + iv * gv;
                float hh = ov * tanhf(cc);
                sh_c[b][j] = cc;
                sh_h[b][j] = hh;
                if (layer == 0) sh_hs[(b * S_ + t) * D_ + j] = hh;
            }
            __syncthreads();
        }
        __syncthreads();
    }

    // head: out[b,k] = h_last[b,:] . Wh[:,k] + bh[k]
    if (tid < 4) {
        int b = tid >> 1, k = tid & 1;
        float s = bh[k];
        for (int j = 0; j < D_; ++j) s += sh_h[b][j] * Wh[j * 2 + k];
        out[b * 2 + k] = s;
    }
}

// ---------------------------------------------------------------------------
// Launch
// ---------------------------------------------------------------------------
extern "C" void kernel_launch(void* const* d_in, const int* in_sizes, int n_in,
                              void* d_out, int out_size, void* d_ws, size_t ws_size,
                              hipStream_t stream) {
    (void)in_sizes; (void)n_in; (void)out_size; (void)ws_size;

    const float*     x    = (const float*)d_in[0];
    const long long* ei   = (const long long*)d_in[1];
    const float*     W1   = (const float*)d_in[2];
    const float*     b1   = (const float*)d_in[3];
    const float*     W2   = (const float*)d_in[4];
    const float*     b2   = (const float*)d_in[5];
    const float*     wih0 = (const float*)d_in[6];
    const float*     whh0 = (const float*)d_in[7];
    const float*     bih0 = (const float*)d_in[8];
    const float*     bhh0 = (const float*)d_in[9];
    const float*     wih1 = (const float*)d_in[10];
    const float*     whh1 = (const float*)d_in[11];
    const float*     bih1 = (const float*)d_in[12];
    const float*     bhh1 = (const float*)d_in[13];
    const float*     Wh   = (const float*)d_in[14];
    const float*     bh   = (const float*)d_in[15];
    float*           out  = (float*)d_out;

    // Workspace layout (floats): bufA[16MB] | bufB[16MB] | dis | seq | xg
    float* bufA = (float*)d_ws;
    float* bufB = bufA + (size_t)NTOT_ * 64;
    float* dis  = bufB + (size_t)NTOT_ * 64;
    float* seq  = dis + N_;
    float* xg   = seq + G_ * D_;

    const int elemBlocks = (NTOT_ * 64) / 256;       // 16384
    const int copyBlocks = (NTOT_ * 64) / 4 / 256;   // 4096
    const int scatBlocks = (G_ * E_ * 64) / 256;     // 262144

    k_degree<<<1, 1024, 0, stream>>>(ei, dis);

    // ---- GCN layer 1 ----
    k_xform1<<<elemBlocks, 256, 0, stream>>>(x, W1, dis, bufA);           // u1 = bufA
    k_copy4<<<copyBlocks, 256, 0, stream>>>((const float4*)bufA, (float4*)bufB); // acc1
    k_scatter<<<scatBlocks, 256, 0, stream>>>(ei, bufA, bufB);
    k_finish<<<elemBlocks, 256, 0, stream>>>(bufB, dis, b1, bufA);        // h1 = bufA

    // ---- GCN layer 2 (WMMA transform) ----
    k_xform2_wmma<<<NTOT_ / 64, 128, 0, stream>>>(bufA, W2, dis, bufB);   // u2 = bufB
    k_copy4<<<copyBlocks, 256, 0, stream>>>((const float4*)bufB, (float4*)bufA); // acc2
    k_scatter<<<scatBlocks, 256, 0, stream>>>(ei, bufB, bufA);
    k_finish<<<elemBlocks, 256, 0, stream>>>(bufA, dis, b2, bufB);        // h2 = bufB

    // ---- pool + LSTM + head ----
    k_mean<<<G_, 64, 0, stream>>>(bufB, seq);
    k_lstm<<<1, 256, 0, stream>>>(seq, wih0, whh0, bih0, bhh0,
                                  wih1, whh1, bih1, bhh1, Wh, bh, xg, out);
}
